// HeterogeneousGAT_31963146617547
// MI455X (gfx1250) — matrix operand order repaired and checked by reading, not compiled, not run on hardware
//
#include <hip/hip_runtime.h>

#define DQ   256   // node feature dim D = H*C
#define ED_  10
#define DIN  20
#define TT   4
#define LL   3
#define HD   64    // pooling head dim
#define AH_  4     // pooling heads

typedef __attribute__((ext_vector_type(16))) _Float16 v16h;
typedef __attribute__((ext_vector_type(8)))  float    v8f;

__device__ __forceinline__ unsigned f2ord(float f){
  unsigned b = __float_as_uint(f);
  return (b & 0x80000000u) ? ~b : (b | 0x80000000u);
}
__device__ __forceinline__ float ord2f(unsigned u){
  return (u & 0x80000000u) ? __uint_as_float(u & 0x7fffffffu) : __uint_as_float(~u);
}
__device__ __forceinline__ float lrelu(float x){ return x > 0.f ? x : 0.2f*x; }

// ---------------------------------------------------------------- utilities
__global__ __launch_bounds__(256) void k_zero(float* p, int n){
  int i = blockIdx.x*256 + threadIdx.x;
  if (i < n) p[i] = 0.f;
}
__global__ __launch_bounds__(256) void k_f32_to_f16(const float* __restrict__ s,
                                                    _Float16* __restrict__ d, int n){
  int i = blockIdx.x*256 + threadIdx.x;
  if (i < n) d[i] = (_Float16)s[i];
}
// src [K x Dd] f32 -> dst [Dd x K] f16
__global__ __launch_bounds__(256) void k_transpose_f16(const float* __restrict__ s,
                                                       _Float16* __restrict__ d, int K, int Dd){
  int o = blockIdx.x*256 + threadIdx.x;
  if (o >= K*Dd) return;
  int dd = o / K, k = o % K;
  d[o] = (_Float16)s[(size_t)k*Dd + dd];
}
__global__ void k_softmax4(const float* __restrict__ w, float* __restrict__ out){
  if (threadIdx.x == 0){
    float m = w[0];
    for (int i=1;i<TT;++i) m = fmaxf(m, w[i]);
    float e[TT], s = 0.f;
    for (int i=0;i<TT;++i){ e[i] = expf(w[i]-m); s += e[i]; }
    for (int i=0;i<TT;++i) out[i] = e[i]/s;
  }
}

// ------------------------------------------------------ WMMA GEMM (f16 -> f32)
// C[M x Nout] = A16[M x K] * Bt16[Nout x K]^T
// One wave computes a 16x64 output strip (4 accumulators): the A fragment is
// loaded once per k-step and reused across 4 WMMAs -> 1.5 vmem ops per wmma.
__global__ __launch_bounds__(256) void k_wmma_gemm(const _Float16* __restrict__ A,
    const _Float16* __restrict__ Bt, float* __restrict__ C, int M, int K, int Nout){
  int wave = (blockIdx.x*256 + threadIdx.x) >> 5;
  int lane = threadIdx.x & 31;
  int ntc  = Nout >> 6;               // 64-wide column strips
  int tr = wave / ntc, tc = wave % ntc;
  if (tr*16 >= M) return;
  int n15 = lane & 15, half = lane >> 4;
  const _Float16* arow = A  + (size_t)(tr*16 + n15)*K;
  const _Float16* b0   = Bt + (size_t)(tc*64 +      n15)*K;
  const _Float16* b1   = Bt + (size_t)(tc*64 + 16 + n15)*K;
  const _Float16* b2   = Bt + (size_t)(tc*64 + 32 + n15)*K;
  const _Float16* b3   = Bt + (size_t)(tc*64 + 48 + n15)*K;
  v8f acc0 = {}, acc1 = {}, acc2 = {}, acc3 = {};
  for (int kb = 0; kb < K; kb += 32){
    const _Float16* ap = arow + kb + half*16;
    __builtin_prefetch(ap + 32, 0, 1);           // warm next k-step of A
    v16h av;
    #pragma unroll
    for (int j=0;j<8;++j){ av[j] = ap[j]; av[8+j] = ap[8+j]; }
    v16h bv;
    #pragma unroll
    for (int j=0;j<16;++j) bv[j] = b0[kb + half*16 + j];
    acc0 = __builtin_amdgcn_wmma_f32_16x16x32_f16(false, av, false, bv, (short)0, acc0, false, false);
    #pragma unroll
    for (int j=0;j<16;++j) bv[j] = b1[kb + half*16 + j];
    acc1 = __builtin_amdgcn_wmma_f32_16x16x32_f16(false, av, false, bv, (short)0, acc1, false, false);
    #pragma unroll
    for (int j=0;j<16;++j) bv[j] = b2[kb + half*16 + j];
    acc2 = __builtin_amdgcn_wmma_f32_16x16x32_f16(false, av, false, bv, (short)0, acc2, false, false);
    #pragma unroll
    for (int j=0;j<16;++j) bv[j] = b3[kb + half*16 + j];
    acc3 = __builtin_amdgcn_wmma_f32_16x16x32_f16(false, av, false, bv, (short)0, acc3, false, false);
  }
  #pragma unroll
  for (int r=0;r<8;++r){
    size_t row = (size_t)(tr*16 + r + 8*half)*Nout + tc*64;
    C[row +      n15] = acc0[r];
    C[row + 16 + n15] = acc1[r];
    C[row + 32 + n15] = acc2[r];
    C[row + 48 + n15] = acc3[r];
  }
}

// ------------------------------------------------------------------ GAT bits
// layer-0: xl = x @ W0t  (K = 20, plain VALU — negligible FLOPs)
__global__ __launch_bounds__(256) void k_lin_small(const float* __restrict__ x,
    const float* __restrict__ W, float* __restrict__ xl, int N){
  int idx = blockIdx.x*256 + threadIdx.x;
  if (idx >= N*DQ) return;
  int n = idx >> 8, d = idx & 255;
  float s = 0.f;
  #pragma unroll
  for (int k=0;k<DIN;++k) s += x[(size_t)n*DIN + k] * W[(size_t)k*DQ + d];
  xl[idx] = s;
}

__global__ __launch_bounds__(256) void k_loop_accum(const int* __restrict__ dst,
    const int* __restrict__ et, int t, const float* __restrict__ eattr,
    float* __restrict__ cnt, float* __restrict__ loopat, int E){
  int e = blockIdx.x*256 + threadIdx.x;
  if (e >= E || et[e] != t) return;
  int d = dst[e];
  atomicAdd(&cnt[d], 1.f);
  #pragma unroll
  for (int j=0;j<ED_;++j) atomicAdd(&loopat[(size_t)d*ED_ + j], eattr[(size_t)e*ED_ + j]);
}
__global__ __launch_bounds__(256) void k_loop_norm(const float* __restrict__ cnt,
    float* __restrict__ loopat, int N){
  int i = blockIdx.x*256 + threadIdx.x;
  if (i >= N*ED_) return;
  loopat[i] /= fmaxf(cnt[i/ED_], 1.f);
}
// we_ae[j] = sum_c We[j][c] * att_edge[c]
__global__ void k_weae(const float* __restrict__ We, const float* __restrict__ ae,
                       float* __restrict__ weae){
  int j = threadIdx.x;
  if (j >= ED_) return;
  float s = 0.f;
  for (int c=0;c<DQ;++c) s += We[(size_t)j*DQ + c]*ae[c];
  weae[j] = s;
}
// per-node: s_src, s_dst (256-dots, wave per node), s_loop, a_loop, smax init
__global__ __launch_bounds__(256) void k_scores(const float* __restrict__ xl,
    const float* __restrict__ loopat, const float* __restrict__ asv,
    const float* __restrict__ adv, const float* __restrict__ weae,
    float* __restrict__ ssrc, float* __restrict__ sdst, float* __restrict__ aloop,
    unsigned* __restrict__ smaxu, int N){
  int wave = blockIdx.x*8 + (threadIdx.x >> 5);
  int lane = threadIdx.x & 31;
  if (wave >= N) return;
  const float* row = xl + (size_t)wave*DQ;
  float ss = 0.f, sd = 0.f;
  for (int j=lane;j<DQ;j+=32){ float v = row[j]; ss += v*asv[j]; sd += v*adv[j]; }
  #pragma unroll
  for (int off=16;off;off>>=1){ ss += __shfl_xor(ss,off,32); sd += __shfl_xor(sd,off,32); }
  if (lane == 0){
    float sl = 0.f;
    #pragma unroll
    for (int j=0;j<ED_;++j) sl += loopat[(size_t)wave*ED_ + j]*weae[j];
    ssrc[wave] = ss; sdst[wave] = sd;
    float a = lrelu(ss + sd + sl);
    aloop[wave] = a;
    smaxu[wave] = f2ord(a);
  }
}
__global__ __launch_bounds__(256) void k_edge_max(const int* __restrict__ src,
    const int* __restrict__ dst, const int* __restrict__ et, int t,
    const float* __restrict__ eattr, const float* __restrict__ weae,
    const float* __restrict__ ssrc, const float* __restrict__ sdst,
    unsigned* __restrict__ smaxu, float* __restrict__ score, int E){
  int e = blockIdx.x*256 + threadIdx.x;
  if (e >= E) return;
  if (et[e] != t){ score[e] = -1e30f; return; }
  float se = 0.f;
  #pragma unroll
  for (int j=0;j<ED_;++j) se += eattr[(size_t)e*ED_ + j]*weae[j];
  float sc = lrelu(ssrc[src[e]] + sdst[dst[e]] + se);
  score[e] = sc;
  atomicMax(&smaxu[dst[e]], f2ord(sc));
}
__global__ __launch_bounds__(256) void k_smax_fin(const unsigned* __restrict__ smaxu,
    const float* __restrict__ aloop, float* __restrict__ smaxf,
    float* __restrict__ pl, float* __restrict__ zsum, int N){
  int n = blockIdx.x*256 + threadIdx.x;
  if (n >= N) return;
  float m = ord2f(smaxu[n]);
  smaxf[n] = m;
  float p = expf(aloop[n] - m);
  pl[n] = p; zsum[n] = p;
}
__global__ __launch_bounds__(256) void k_edge_p(const int* __restrict__ dst,
    const int* __restrict__ et, int t, float* __restrict__ score,
    const float* __restrict__ smaxf, float* __restrict__ zsum, int E){
  int e = blockIdx.x*256 + threadIdx.x;
  if (e >= E) return;
  if (et[e] != t){ score[e] = 0.f; return; }
  float p = expf(score[e] - smaxf[dst[e]]);
  score[e] = p;
  atomicAdd(&zsum[dst[e]], p);
}
__global__ __launch_bounds__(256) void k_agg_self(const float* __restrict__ xl,
    const float* __restrict__ pl, const float* __restrict__ zsum,
    float* __restrict__ outagg, int N){
  int idx = blockIdx.x*256 + threadIdx.x;
  if (idx >= N*DQ) return;
  int n = idx >> 8;
  outagg[idx] = (pl[n]/(zsum[n] + 1e-16f)) * xl[idx];
}
__global__ __launch_bounds__(256) void k_agg_edge(const int* __restrict__ src,
    const int* __restrict__ dst, const int* __restrict__ et, int t,
    const float* __restrict__ score, const float* __restrict__ zsum,
    const float* __restrict__ xl, float* __restrict__ outagg){
  int e = blockIdx.x;
  if (et[e] != t) return;
  int d = threadIdx.x;
  int de = dst[e], se = src[e];
  float alpha = score[e]/(zsum[de] + 1e-16f);
  atomicAdd(&outagg[(size_t)de*DQ + d], alpha * xl[(size_t)se*DQ + d]);
}
__global__ __launch_bounds__(256) void k_finish(const float* __restrict__ outagg,
    const float* __restrict__ bias, float* __restrict__ h, int N){
  int idx = blockIdx.x*256 + threadIdx.x;
  if (idx >= N*DQ) return;
  float v = outagg[idx] + bias[idx & 255];
  h[idx] = v > 0.f ? v : expm1f(v);
}
__global__ __launch_bounds__(256) void k_axpy(float* __restrict__ acc,
    const float* __restrict__ h, const float* __restrict__ wbuf, int t, int N){
  int idx = blockIdx.x*256 + threadIdx.x;
  if (idx < N*DQ) acc[idx] += wbuf[t] * h[idx];
}

// ---------------------------------------------------------------- pooling
__global__ __launch_bounds__(256) void k_mean_max(const float* __restrict__ h,
    float* __restrict__ meanp, float* __restrict__ maxp, int N){
  int d = blockIdx.x, tid = threadIdx.x;
  float s = 0.f, m = -1e30f;
  for (int n=tid;n<N;n+=256){ float v = h[(size_t)n*DQ + d]; s += v; m = fmaxf(m, v); }
  __shared__ float ss[256], sm[256];
  ss[tid] = s; sm[tid] = m; __syncthreads();
  for (int off=128;off;off>>=1){
    if (tid < off){ ss[tid] += ss[tid+off]; sm[tid] = fmaxf(sm[tid], sm[tid+off]); }
    __syncthreads();
  }
  if (tid == 0){ meanp[d] = ss[0]/(float)N; maxp[d] = sm[0]; }
}
__global__ __launch_bounds__(256) void k_qkv_split(const float* __restrict__ qkvf,
    const float* __restrict__ ipb, _Float16* __restrict__ q,
    _Float16* __restrict__ k, _Float16* __restrict__ v, int N){
  int idx = blockIdx.x*256 + threadIdx.x;
  if (idx >= N*3*DQ) return;
  int n = idx/(3*DQ), o = idx%(3*DQ);
  float val = qkvf[idx] + ipb[o];
  if (o < DQ)        q[(size_t)n*DQ + o]        = (_Float16)val;
  else if (o < 2*DQ) k[(size_t)n*DQ + o - DQ]   = (_Float16)val;
  else               v[(size_t)n*DQ + o - 2*DQ] = (_Float16)val;
}

// flash attention: one wave per (head, 16-query tile); accumulates column sums
__global__ __launch_bounds__(32) void k_flash(const _Float16* __restrict__ q,
    const _Float16* __restrict__ k, const _Float16* __restrict__ v,
    float* __restrict__ sumO, int N){
  int rt = blockIdx.x, h = blockIdx.y;
  int lane = threadIdx.x, n15 = lane & 15, half = lane >> 4;
  const int hoff = h*HD;

  const _Float16* qrow = q + (size_t)(rt*16 + n15)*DQ + hoff;
  v16h aq0, aq1;
  #pragma unroll
  for (int j=0;j<8;++j){
    aq0[j]   = qrow[half*16 + j];
    aq0[8+j] = qrow[8 + half*16 + j];
    aq1[j]   = qrow[32 + half*16 + j];
    aq1[8+j] = qrow[40 + half*16 + j];
  }
  float rmax[8], rsum[8];
  #pragma unroll
  for (int r=0;r<8;++r){ rmax[r] = -1e30f; rsum[r] = 0.f; }
  v8f o0 = {}, o1 = {}, o2 = {}, o3 = {};

  __shared__ _Float16 pbuf[16*32];
  __shared__ _Float16 vbuf[32*64];

  for (int kb=0; kb<N; kb+=32){
    v8f s0 = {}, s1 = {};
    {
      const _Float16* kr0 = k + (size_t)(kb + n15)*DQ + hoff;
      const _Float16* kr1 = k + (size_t)(kb + 16 + n15)*DQ + hoff;
      v16h b;
      #pragma unroll
      for (int j=0;j<16;++j) b[j] = kr0[half*16 + j];
      s0 = __builtin_amdgcn_wmma_f32_16x16x32_f16(false, aq0, false, b, (short)0, s0, false, false);
      #pragma unroll
      for (int j=0;j<16;++j) b[j] = kr0[32 + half*16 + j];
      s0 = __builtin_amdgcn_wmma_f32_16x16x32_f16(false, aq1, false, b, (short)0, s0, false, false);
      #pragma unroll
      for (int j=0;j<16;++j) b[j] = kr1[half*16 + j];
      s1 = __builtin_amdgcn_wmma_f32_16x16x32_f16(false, aq0, false, b, (short)0, s1, false, false);
      #pragma unroll
      for (int j=0;j<16;++j) b[j] = kr1[32 + half*16 + j];
      s1 = __builtin_amdgcn_wmma_f32_16x16x32_f16(false, aq1, false, b, (short)0, s1, false, false);
    }
    { // stage V tile (32 keys x 64 dims), one row per lane, coalesced
      const _Float16* vr = v + (size_t)(kb + lane)*DQ + hoff;
      #pragma unroll
      for (int j=0;j<64;++j) vbuf[lane*64 + j] = vr[j];
    }
    #pragma unroll
    for (int r=0;r<8;++r){
      float a0 = s0[r]*0.125f, a1 = s1[r]*0.125f;   // 1/sqrt(64)
      float mv = fmaxf(a0, a1);
      #pragma unroll
      for (int off=1; off<16; off<<=1) mv = fmaxf(mv, __shfl_xor(mv, off, 32));
      float mn  = fmaxf(rmax[r], mv);
      float fac = __expf(rmax[r] - mn);
      rmax[r] = mn;
      float p0 = __expf(a0 - mn), p1 = __expf(a1 - mn);
      float ps = p0 + p1;
      #pragma unroll
      for (int off=1; off<16; off<<=1) ps += __shfl_xor(ps, off, 32);
      rsum[r] = rsum[r]*fac + ps;
      o0[r]*=fac; o1[r]*=fac; o2[r]*=fac; o3[r]*=fac;
      pbuf[(r + 8*half)*32 + n15]      = (_Float16)p0;
      pbuf[(r + 8*half)*32 + 16 + n15] = (_Float16)p1;
    }
    __syncthreads();
    v16h ap;
    #pragma unroll
    for (int j=0;j<8;++j){
      ap[j]   = pbuf[n15*32 + half*16 + j];
      ap[8+j] = pbuf[n15*32 + 8 + half*16 + j];
    }
    v16h bv;
    #pragma unroll
    for (int j=0;j<16;++j) bv[j] = vbuf[(half*16 + j)*64 + n15];
    o0 = __builtin_amdgcn_wmma_f32_16x16x32_f16(false, ap, false, bv, (short)0, o0, false, false);
    #pragma unroll
    for (int j=0;j<16;++j) bv[j] = vbuf[(half*16 + j)*64 + 16 + n15];
    o1 = __builtin_amdgcn_wmma_f32_16x16x32_f16(false, ap, false, bv, (short)0, o1, false, false);
    #pragma unroll
    for (int j=0;j<16;++j) bv[j] = vbuf[(half*16 + j)*64 + 32 + n15];
    o2 = __builtin_amdgcn_wmma_f32_16x16x32_f16(false, ap, false, bv, (short)0, o2, false, false);
    #pragma unroll
    for (int j=0;j<16;++j) bv[j] = vbuf[(half*16 + j)*64 + 48 + n15];
    o3 = __builtin_amdgcn_wmma_f32_16x16x32_f16(false, ap, false, bv, (short)0, o3, false, false);
    __syncthreads();
  }
  float c0=0.f, c1=0.f, c2=0.f, c3=0.f;
  #pragma unroll
  for (int r=0;r<8;++r){
    float z = 1.f/rsum[r];
    c0 += o0[r]*z; c1 += o1[r]*z; c2 += o2[r]*z; c3 += o3[r]*z;
  }
  atomicAdd(&sumO[hoff + n15],      c0);
  atomicAdd(&sumO[hoff + 16 + n15], c1);
  atomicAdd(&sumO[hoff + 32 + n15], c2);
  atomicAdd(&sumO[hoff + 48 + n15], c3);
}

// out_proj on attention mean + concat + fusion MLP -> d_out[256]
__global__ __launch_bounds__(256) void k_fuse(const float* __restrict__ meanp,
    const float* __restrict__ maxp, const float* __restrict__ sumO,
    const float* __restrict__ opw, const float* __restrict__ opb,
    const float* __restrict__ fw1, const float* __restrict__ fb1,
    const float* __restrict__ fw2, const float* __restrict__ fb2,
    float* __restrict__ out, int N){
  __shared__ float comb[3*DQ];
  __shared__ float g1s[2*DQ];
  int tid = threadIdx.x;
  float inv = 1.f/(float)N;
  float s = opb[tid];
  for (int kk=0;kk<DQ;++kk) s += (sumO[kk]*inv) * opw[(size_t)tid*DQ + kk];
  comb[tid]        = meanp[tid];
  comb[DQ + tid]   = maxp[tid];
  comb[2*DQ + tid] = s;
  __syncthreads();
  for (int j=tid;j<2*DQ;j+=256){
    float a = fb1[j];
    for (int i=0;i<3*DQ;++i) a += comb[i]*fw1[(size_t)i*(2*DQ) + j];
    g1s[j] = a > 0.f ? a : 0.f;
  }
  __syncthreads();
  float a = fb2[tid];
  for (int i=0;i<2*DQ;++i) a += g1s[i]*fw2[(size_t)i*DQ + tid];
  out[tid] = a > 0.f ? a : 0.f;
}

// ----------------------------------------------------------------- host
static inline void* wsa(char*& p, size_t bytes){
  void* r = (void*)p;
  p += (bytes + 255) & ~(size_t)255;
  return r;
}

extern "C" void kernel_launch(void* const* d_in, const int* in_sizes, int n_in,
                              void* d_out, int out_size, void* d_ws, size_t ws_size,
                              hipStream_t stream) {
  const float* x        = (const float*)d_in[0];
  const int*   srcv     = (const int*)  d_in[1];
  const int*   dstv     = (const int*)  d_in[2];
  const float* eattr    = (const float*)d_in[3];
  const int*   etype    = (const int*)  d_in[4];
  const float* W0       = (const float*)d_in[5];
  const float* W12      = (const float*)d_in[6];
  const float* att_src  = (const float*)d_in[7];
  const float* att_dst  = (const float*)d_in[8];
  const float* att_edge = (const float*)d_in[9];
  const float* lin_ew   = (const float*)d_in[10];
  const float* biases   = (const float*)d_in[11];
  const float* etw      = (const float*)d_in[12];
  const float* ipw      = (const float*)d_in[13];
  const float* ipb      = (const float*)d_in[14];
  const float* opw      = (const float*)d_in[15];
  const float* opb      = (const float*)d_in[16];
  const float* fw1      = (const float*)d_in[17];
  const float* fb1      = (const float*)d_in[18];
  const float* fw2      = (const float*)d_in[19];
  const float* fb2      = (const float*)d_in[20];
  (void)n_in; (void)out_size; (void)ws_size;

  const int N = in_sizes[0] / DIN;   // 4096
  const int E = in_sizes[1];         // 65536
  const int ND = N*DQ;

  char* p = (char*)d_ws;
  float*    h_cur  = (float*)   wsa(p, (size_t)ND*4);
  float*    h_nxt  = (float*)   wsa(p, (size_t)ND*4);
  float*    h_comb = (float*)   wsa(p, (size_t)ND*4);
  float*    xl     = (float*)   wsa(p, (size_t)ND*4);
  float*    outagg = (float*)   wsa(p, (size_t)ND*4);
  float*    qkvf   = (float*)   wsa(p, (size_t)N*3*DQ*4);
  _Float16* a16    = (_Float16*)wsa(p, (size_t)ND*2);
  _Float16* b16    = (_Float16*)wsa(p, (size_t)3*DQ*DQ*2);
  _Float16* q16    = (_Float16*)wsa(p, (size_t)ND*2);
  _Float16* k16    = (_Float16*)wsa(p, (size_t)ND*2);
  _Float16* v16    = (_Float16*)wsa(p, (size_t)ND*2);
  float*    cnt    = (float*)   wsa(p, (size_t)N*4);
  float*    loopat = (float*)   wsa(p, (size_t)N*ED_*4);
  float*    s_src  = (float*)   wsa(p, (size_t)N*4);
  float*    s_dst  = (float*)   wsa(p, (size_t)N*4);
  float*    a_loop = (float*)   wsa(p, (size_t)N*4);
  unsigned* smaxu  = (unsigned*)wsa(p, (size_t)N*4);
  float*    smaxf  = (float*)   wsa(p, (size_t)N*4);
  float*    pl     = (float*)   wsa(p, (size_t)N*4);
  float*    zsum   = (float*)   wsa(p, (size_t)N*4);
  float*    scoreb = (float*)   wsa(p, (size_t)E*4);
  float*    weae   = (float*)   wsa(p, 64);
  float*    ws4    = (float*)   wsa(p, 64);
  float*    sumO   = (float*)   wsa(p, (size_t)DQ*4);
  float*    meanp  = (float*)   wsa(p, (size_t)DQ*4);
  float*    maxp   = (float*)   wsa(p, (size_t)DQ*4);

  const int gN   = (N + 255)/256;
  const int gND  = (ND + 255)/256;
  const int gE   = (E + 255)/256;
  const int gN10 = (N*ED_ + 255)/256;

  k_softmax4<<<1, 32, 0, stream>>>(etw, ws4);
  k_zero<<<gND, 256, 0, stream>>>(h_comb, ND);

  for (int t = 0; t < TT; ++t){
    // per-type self-loop edge_attr = mean of masked incoming edge_attr
    k_zero<<<gN,   256, 0, stream>>>(cnt, N);
    k_zero<<<gN10, 256, 0, stream>>>(loopat, N*ED_);
    k_loop_accum<<<gE, 256, 0, stream>>>(dstv, etype, t, eattr, cnt, loopat, E);
    k_loop_norm<<<gN10, 256, 0, stream>>>(cnt, loopat, N);

    float* hA = h_cur; float* hB = h_nxt;
    for (int i = 0; i < LL; ++i){
      if (i == 0){
        k_lin_small<<<gND, 256, 0, stream>>>(x, W0 + (size_t)t*DIN*DQ, xl, N);
      } else {
        k_f32_to_f16<<<gND, 256, 0, stream>>>(hA, a16, ND);
        k_transpose_f16<<<(DQ*DQ)/256, 256, 0, stream>>>(
            W12 + ((size_t)(t*2 + (i-1)))*DQ*DQ, b16, DQ, DQ);
        int tiles = (N/16)*(DQ/64);          // 16x64 strip per wave
        k_wmma_gemm<<<tiles/8, 256, 0, stream>>>(a16, b16, xl, N, DQ, DQ);
      }
      size_t sl = (size_t)(t*LL + i)*DQ;
      k_weae<<<1, 32, 0, stream>>>(lin_ew + (size_t)(t*LL + i)*ED_*DQ, att_edge + sl, weae);
      k_scores<<<N/8, 256, 0, stream>>>(xl, loopat, att_src + sl, att_dst + sl, weae,
                                        s_src, s_dst, a_loop, smaxu, N);
      k_edge_max<<<gE, 256, 0, stream>>>(srcv, dstv, etype, t, eattr, weae,
                                         s_src, s_dst, smaxu, scoreb, E);
      k_smax_fin<<<gN, 256, 0, stream>>>(smaxu, a_loop, smaxf, pl, zsum, N);
      k_edge_p<<<gE, 256, 0, stream>>>(dstv, etype, t, scoreb, smaxf, zsum, E);
      k_agg_self<<<gND, 256, 0, stream>>>(xl, pl, zsum, outagg, N);
      k_agg_edge<<<E, 256, 0, stream>>>(srcv, dstv, etype, t, scoreb, zsum, xl, outagg);
      k_finish<<<gND, 256, 0, stream>>>(outagg, biases + sl, hB, N);
      float* tmp = hA; hA = hB; hB = tmp;
    }
    k_axpy<<<gND, 256, 0, stream>>>(h_comb, hA, ws4, t, N);
  }

  // ---- hierarchical pooling ----
  k_mean_max<<<DQ, 256, 0, stream>>>(h_comb, meanp, maxp, N);

  k_f32_to_f16<<<gND, 256, 0, stream>>>(h_comb, a16, ND);
  k_f32_to_f16<<<(3*DQ*DQ)/256, 256, 0, stream>>>(ipw, b16, 3*DQ*DQ);  // already [out][k]
  {
    int tiles = (N/16)*((3*DQ)/64);          // 16x64 strip per wave
    k_wmma_gemm<<<tiles/8, 256, 0, stream>>>(a16, b16, qkvf, N, DQ, 3*DQ);
  }
  k_qkv_split<<<(N*3*DQ)/256, 256, 0, stream>>>(qkvf, ipb, q16, k16, v16, N);

  k_zero<<<1, 256, 0, stream>>>(sumO, DQ);
  {
    dim3 fg(N/16, AH_);
    k_flash<<<fg, 32, 0, stream>>>(q16, k16, v16, sumO, N);
  }
  k_fuse<<<1, 256, 0, stream>>>(meanp, maxp, sumO, opw, opb,
                                fw1, fb1, fw2, fb2, (float*)d_out, N);
}
